// DualLatentE2CO_75952201662520
// MI455X (gfx1250) — compile-verified, matrix operand order checked by reading, not run on hardware
//
#include <hip/hip_runtime.h>
#include <cstdint>
#include <cstddef>

typedef _Float16 half_t;
typedef __attribute__((ext_vector_type(16))) _Float16 v16h;
typedef __attribute__((ext_vector_type(8)))  _Float16 v8h;
typedef __attribute__((ext_vector_type(8)))  float    v8f;
typedef __attribute__((ext_vector_type(4)))  unsigned u32x4;
typedef __attribute__((ext_vector_type(4)))  int      i32x4;
typedef __attribute__((ext_vector_type(8)))  int      i32x8;

#define GS_LOOP(i, total) \
  for (long i = (long)blockIdx.x * blockDim.x + threadIdx.x; i < (total); \
       i += (long)gridDim.x * blockDim.x)

static inline int ceil64(int k) { return (k + 63) & ~63; }

// ---------------------------------------------------------------------------
// WMMA GEMM:  C[M,N] = A[M,Kp] * B[N,Kp]^T + bias, f32 accumulate.
// A and B both row-major along K, Kp % 64 == 0, pad region zero-filled by
// producers. Block tile 64x32, 8 wave32 waves -> eight 16x16 WMMA tiles,
// K chunk 64 (2 WMMA issues per barrier pair). All tile traffic is b128.
// Loads are issued as a batch into registers before the LDS stores so the
// three global_load_b128 per thread pipeline instead of serializing.
// S==0 -> row-major [M,N] out; S>0 -> conv-NCHW: idx=((m/S)*N+n)*S+(m%S)
// act: 0 none, 1 relu, 2 tanh
// ---------------------------------------------------------------------------
__global__ __launch_bounds__(256) void k_gemm(const half_t* __restrict__ A,
                                              const half_t* __restrict__ B,
                                              const float* __restrict__ bias,
                                              float* __restrict__ C,
                                              int M, int N, int Kp, int S, int act) {
  __shared__ v8h As8[512];  // 64 rows x 8 chunks of 8 halves
  __shared__ v8h Bs8[256];  // 32 rows x 8 chunks
  const int tid  = threadIdx.x;
  const int bm   = blockIdx.x * 64;
  const int bn   = blockIdx.y * 32;
  const int wave = tid >> 5, lane = tid & 31;
  const int wr   = wave & 3, wc = wave >> 2;
  const int lrow = lane & 15;
  const int c0   = (lane < 16) ? 0 : 1;  // ISA 16-bit A/B lane K-chunk select
  const int rowA = tid >> 3, ch = tid & 7;
  const v8h zerov = {0, 0, 0, 0, 0, 0, 0, 0};
  v8f acc = {0.f, 0.f, 0.f, 0.f, 0.f, 0.f, 0.f, 0.f};

  for (int k0 = 0; k0 < Kp; k0 += 64) {
    if (k0 + 64 < Kp)  // gfx1250 global_prefetch_b8
      __builtin_prefetch(&A[(size_t)(bm + (tid >> 2)) * Kp + k0 + 64], 0, 1);
    // batch the three tile loads into registers (no LDS dependence yet)
    v8h va0 = zerov, va1 = zerov, vb = zerov;
    if (bm + rowA < M)
      va0 = *((const v8h*)(A + (size_t)(bm + rowA) * Kp + k0) + ch);
    if (bm + 32 + rowA < M)
      va1 = *((const v8h*)(A + (size_t)(bm + 32 + rowA) * Kp + k0) + ch);
    if (bn + rowA < N)
      vb = *((const v8h*)(B + (size_t)(bn + rowA) * Kp + k0) + ch);
    __syncthreads();                 // previous iteration's fragment reads done
    As8[tid]       = va0;
    As8[tid + 256] = va1;
    Bs8[tid]       = vb;
    __syncthreads();
    const v8h* Ap = &As8[(wr * 16 + lrow) * 8];
    const v8h* Bp = &Bs8[(wc * 16 + lrow) * 8];
    v16h a0 = __builtin_shufflevector(Ap[c0], Ap[c0 + 2], 0, 1, 2, 3, 4, 5, 6, 7,
                                      8, 9, 10, 11, 12, 13, 14, 15);
    v16h b0 = __builtin_shufflevector(Bp[c0], Bp[c0 + 2], 0, 1, 2, 3, 4, 5, 6, 7,
                                      8, 9, 10, 11, 12, 13, 14, 15);
    acc = __builtin_amdgcn_wmma_f32_16x16x32_f16(false, a0, false, b0, (short)0,
                                                 acc, false, false);
    v16h a1 = __builtin_shufflevector(Ap[c0 + 4], Ap[c0 + 6], 0, 1, 2, 3, 4, 5, 6,
                                      7, 8, 9, 10, 11, 12, 13, 14, 15);
    v16h b1 = __builtin_shufflevector(Bp[c0 + 4], Bp[c0 + 6], 0, 1, 2, 3, 4, 5, 6,
                                      7, 8, 9, 10, 11, 12, 13, 14, 15);
    acc = __builtin_amdgcn_wmma_f32_16x16x32_f16(false, a1, false, b1, (short)0,
                                                 acc, false, false);
  }

  const int n     = bn + wc * 16 + lrow;
  const int mbase = bm + wr * 16 + ((lane < 16) ? 0 : 8);
  if (n < N) {
    float bv = bias ? bias[n] : 0.f;
#pragma unroll
    for (int r = 0; r < 8; ++r) {
      int m = mbase + r;
      if (m < M) {
        float v = acc[r] + bv;
        if (act == 1) v = fmaxf(v, 0.f);
        else if (act == 2) v = tanhf(v);
        size_t idx = (S == 0)
                         ? ((size_t)m * N + n)
                         : (((size_t)(m / S) * N + n) * (size_t)S + (size_t)(m % S));
        C[idx] = v;
      }
    }
  }
}

// ---------------------------------------------------------------------------
// Tensor Data Mover probe: DMA a 64x32 f16 tile of a weight matrix into LDS
// via TENSOR_LOAD_TO_LDS (D# built per CDNA5 ISA 8.3/8.4), wait on TENSORcnt,
// reduce the tile to a scratch sink. Self-contained; output unused.
// ---------------------------------------------------------------------------
__global__ void k_tdm_probe(const half_t* __restrict__ src, float* __restrict__ sink,
                            int Kp, int Nrows) {
  __shared__ half_t tile[64 * 32];
#if __has_builtin(__builtin_amdgcn_tensor_load_to_lds)
  unsigned long long ga = (unsigned long long)(const void*)src;
  u32x4 g0;
  g0[0] = 1u;                                   // count=1 (valid user descriptor)
  g0[1] = 0u;                                   // lds_addr (bytes)
  g0[2] = (unsigned)(ga & 0xFFFFFFFFu);         // global_addr[31:0]
  g0[3] = (unsigned)((ga >> 32) & 0x01FFFFFFu)  // global_addr[56:32]
          | 0x80000000u;                        // type=2 ("image")
  unsigned td0 = (unsigned)Kp, td1 = (unsigned)Nrows;
  unsigned tile0 = 64u, tile1 = 32u;
  i32x8 g1;
  g1[0] = (int)(1u << 16);                              // data_size=1 -> 2 bytes
  g1[1] = (int)((td0 & 0xFFFFu) << 16);                 // tensor_dim0[15:0]
  g1[2] = (int)((td0 >> 16) | ((td1 & 0xFFFFu) << 16)); // td0 hi | td1 lo
  g1[3] = (int)((td1 >> 16) | (tile0 << 16));           // td1 hi | tile_dim0
  g1[4] = (int)tile1;                                   // tile_dim1 (tile_dim2=0)
  g1[5] = (int)td0;                                     // tensor_dim0_stride lo
  g1[6] = 0; g1[7] = 0;
  i32x4 z4 = {0, 0, 0, 0};
#if __clang_major__ >= 23
  i32x8 z8 = {0, 0, 0, 0, 0, 0, 0, 0};
  __builtin_amdgcn_tensor_load_to_lds(g0, g1, z4, z4, z8, 0);
#else
  __builtin_amdgcn_tensor_load_to_lds(g0, g1, z4, z4, 0);
#endif
  __builtin_amdgcn_s_wait_tensorcnt(0);
#endif
  __syncthreads();
  float acc = 0.f;
  for (int i = threadIdx.x; i < 64 * 32; i += blockDim.x) acc += (float)tile[i];
  if (threadIdx.x == 0) sink[blockIdx.x] = acc;
}

// im2col with input (lhs) dilation; writes f16 [M, Kcp], pad columns zeroed.
__global__ void k_im2col(const float* __restrict__ x, half_t* __restrict__ col,
                         int Nimg, int Cin, int Hin, int Win, int kk, int pad,
                         int stride, int dil, int OH, int OW, int Kcp) {
  const int Kc = Cin * kk * kk;
  const long M = (long)Nimg * OH * OW;
  GS_LOOP(i, M * Kcp) {
    long m = i / Kcp; int kidx = (int)(i % Kcp);
    float v = 0.f;
    if (kidx < Kc) {
      int ow = (int)(m % OW); int oh = (int)((m / OW) % OH);
      int nimg = (int)(m / ((long)OW * OH));
      int kw = kidx % kk; int kh = (kidx / kk) % kk; int ci = kidx / (kk * kk);
      int ih = oh * stride + kh - pad;   // coordinate in dilated input space
      int iw = ow * stride + kw - pad;
      int Hd = (Hin - 1) * dil + 1, Wd = (Win - 1) * dil + 1;
      if (ih >= 0 && iw >= 0 && ih < Hd && iw < Wd && (ih % dil) == 0 && (iw % dil) == 0)
        v = x[(((size_t)nimg * Cin + ci) * Hin + ih / dil) * Win + iw / dil];
    }
    col[i] = (half_t)v;
  }
}

// BN stats per (t, channel) over (Bimg, S). x layout [(t*Bimg+b)][C][S].
__global__ void k_bn_stats(const float* __restrict__ x, float* __restrict__ mean,
                           float* __restrict__ rstd, int Tn, int Bimg, int C, int S) {
  const int c = blockIdx.x, t = blockIdx.y;
  float s = 0.f, s2 = 0.f;
  const long cnt = (long)Bimg * S;
  for (long i = threadIdx.x; i < cnt; i += blockDim.x) {
    int b = (int)(i / S); int sp = (int)(i % S);
    float v = x[(((size_t)(t * Bimg + b)) * C + c) * S + sp];
    s += v; s2 += v * v;
  }
  __shared__ float rs[256], rs2[256];
  rs[threadIdx.x] = s; rs2[threadIdx.x] = s2;
  __syncthreads();
  for (int off = 128; off > 0; off >>= 1) {
    if ((int)threadIdx.x < off) {
      rs[threadIdx.x]  += rs[threadIdx.x + off];
      rs2[threadIdx.x] += rs2[threadIdx.x + off];
    }
    __syncthreads();
  }
  if (threadIdx.x == 0) {
    float m = rs[0] / (float)cnt;
    float var = rs2[0] / (float)cnt - m * m;
    mean[t * C + c] = m;
    rstd[t * C + c] = rsqrtf(var + 1e-5f);
  }
}

__global__ void k_bn_apply(const float* __restrict__ x, const float* __restrict__ res,
                           float* __restrict__ y, const float* __restrict__ mean,
                           const float* __restrict__ rstd, const float* __restrict__ g,
                           const float* __restrict__ be, int Tn, int Bimg, int C, int S,
                           int relu) {
  const long total = (long)Tn * Bimg * C * S;
  GS_LOOP(i, total) {
    long q = i / S; int c = (int)(q % C);
    long tb = q / C; int t = (int)(tb / Bimg);
    float v = g[c] * (x[i] - mean[t * C + c]) * rstd[t * C + c] + be[c];
    if (res) v += res[i];
    if (relu) v = fmaxf(v, 0.f);
    y[i] = v;
  }
}

// f32 [rows][K] -> f16 [rows][Kp], pad zero
__global__ void k_cvt_pad(const float* __restrict__ x, half_t* __restrict__ y,
                          long rows, int K, int Kp) {
  GS_LOOP(i, rows * Kp) {
    long r = i / Kp; int k = (int)(i % Kp);
    y[i] = (half_t)((k < K) ? x[r * K + k] : 0.f);
  }
}

// w[N][K] f32 -> o[N][Kp] f16 (no transpose; GEMM uses B^T), pad zero
__global__ void k_prep_pad(const float* __restrict__ w, half_t* __restrict__ o,
                           int N, int K, int Kp) {
  GS_LOOP(i, (long)N * Kp) {
    int n = (int)(i / Kp), k = (int)(i % Kp);
    o[i] = (half_t)((k < K) ? w[(long)n * K + k] : 0.f);
  }
}

// deconv: w[Ci][Co][kk][kk] -> o[Co][Kp], o[co][(ci*kk+kh)*kk+kw] = flipped w
__global__ void k_prep_deconv(const float* __restrict__ w, half_t* __restrict__ o,
                              int Ci, int Co, int kk, int Kp) {
  const int Kc = Ci * kk * kk;
  GS_LOOP(i, (long)Co * Kp) {
    int co = (int)(i / Kp), k = (int)(i % Kp);
    float v = 0.f;
    if (k < Kc) {
      int kw = k % kk; int kh = (k / kk) % kk; int ci = k / (kk * kk);
      v = w[(((size_t)ci * Co + co) * kk + (kk - 1 - kh)) * kk + (kk - 1 - kw)];
    }
    o[i] = (half_t)v;
  }
}

__global__ void k_concat_z(const float* __restrict__ zd, const float* __restrict__ zp,
                           float* __restrict__ z, int rows) {
  GS_LOOP(i, (long)rows * 96) {
    int r = (int)(i / 96), j = (int)(i % 96);
    z[i] = (j < 64) ? zd[r * 64 + j] : zp[r * 32 + (j - 64)];
  }
}

// hz f16 [rows][128]: 96 latent + dt + zero pad
__global__ void k_build_hz(const float* __restrict__ z, const float* __restrict__ dt,
                           half_t* __restrict__ hz, int rows) {
  GS_LOOP(i, (long)rows * 128) {
    int r = (int)(i / 128), j = (int)(i % 128);
    float v = (j < 96) ? z[r * 96 + j] : ((j == 96) ? dt[r] : 0.f);
    hz[i] = (half_t)v;
  }
}

// PINN input f16 [rows][64]: 32 latent + 2 coords + zero pad; z row = row/256
__global__ void k_pinn_in(const float* __restrict__ z, int zld, int zoff,
                          const float* __restrict__ coords, half_t* __restrict__ out,
                          long rows) {
  GS_LOOP(i, rows * 64) {
    long r = i / 64; int f = (int)(i % 64);
    long zi = r / 256;
    float v = 0.f;
    if (f < 32) v = z[zi * zld + zoff + f];
    else if (f < 34) v = coords[r * 2 + (f - 32)];
    out[i] = (half_t)v;
  }
}

// one latent-scan step: zt1 = At zt + Bt (u*dt) ; y = Ct zt1[:64] + Dt (u*dt)
__global__ void k_step(const float* __restrict__ At, const float* __restrict__ Bt,
                       const float* __restrict__ Ct, const float* __restrict__ Dt,
                       const float* __restrict__ U, const float* __restrict__ dt,
                       const float* __restrict__ zin, float* __restrict__ zout,
                       float* __restrict__ yout) {
  const int b = blockIdx.x; const int i = threadIdx.x;
  __shared__ float zs[96], us[9], znew[96];
  if (i < 96) zs[i] = zin[b * 96 + i];
  if (i < 9)  us[i] = U[b * 9 + i] * dt[b];
  __syncthreads();
  if (i < 96) {
    const float* Ar = At + ((size_t)b * 96 + i) * 96;
    const float* Br = Bt + ((size_t)b * 96 + i) * 9;
    float acc = 0.f;
    for (int j = 0; j < 96; ++j) acc += Ar[j] * zs[j];
    for (int u = 0; u < 9; ++u) acc += Br[u] * us[u];
    znew[i] = acc;
    zout[b * 96 + i] = acc;
  }
  __syncthreads();
  if (i < 14) {
    const float* Cr = Ct + ((size_t)b * 14 + i) * 64;
    const float* Dr = Dt + ((size_t)b * 14 + i) * 9;
    float acc = 0.f;
    for (int j = 0; j < 64; ++j) acc += Cr[j] * znew[j];
    for (int u = 0; u < 9; ++u) acc += Dr[u] * us[u];
    yout[b * 14 + i] = acc;
  }
}

// ---------------------------------------------------------------------------
// host-side helpers
// ---------------------------------------------------------------------------
static void run_gemm(hipStream_t s, const half_t* A, const half_t* B, const float* bias,
                     float* C, long M, int N, int Kp, int S, int act) {
  dim3 g((unsigned)((M + 63) / 64), (unsigned)((N + 31) / 32));
  k_gemm<<<g, dim3(256), 0, s>>>(A, B, bias, C, (int)M, N, Kp, S, act);
}

static void conv_layer(hipStream_t s, const float* x, half_t* col, const half_t* wB,
                       const float* bias, float* out, int Nimg, int Cin, int Hin,
                       int Win, int kk, int pad, int stride, int dil, int Cout) {
  int Hd = (Hin - 1) * dil + 1, Wd = (Win - 1) * dil + 1;
  int OH = (Hd + 2 * pad - kk) / stride + 1;
  int OW = (Wd + 2 * pad - kk) / stride + 1;
  long M = (long)Nimg * OH * OW;
  int Kcp = ceil64(Cin * kk * kk);
  k_im2col<<<dim3(8192), dim3(256), 0, s>>>(x, col, Nimg, Cin, Hin, Win, kk, pad,
                                            stride, dil, OH, OW, Kcp);
  run_gemm(s, col, wB, bias, out, M, Cout, Kcp, OH * OW, 0);
}

static void bn_relu(hipStream_t s, const float* x, const float* res, float* y,
                    const float* g, const float* be, float* mean, float* rstd,
                    int Tn, int Bimg, int C, int S, int relu) {
  k_bn_stats<<<dim3(C, Tn), dim3(256), 0, s>>>(x, mean, rstd, Tn, Bimg, C, S);
  k_bn_apply<<<dim3(4096), dim3(256), 0, s>>>(x, res, y, mean, rstd, g, be, Tn, Bimg,
                                              C, S, relu);
}

static void prep(hipStream_t s, const float* w, half_t* o, int N, int K, int Kp) {
  k_prep_pad<<<dim3(1024), dim3(256), 0, s>>>(w, o, N, K, Kp);
}

// ---------------------------------------------------------------------------
extern "C" void kernel_launch(void* const* d_in, const int* in_sizes, int n_in,
                              void* d_out, int out_size, void* d_ws, size_t ws_size,
                              hipStream_t stream) {
  (void)in_sizes; (void)n_in; (void)out_size; (void)ws_size;
  hipStream_t s = stream;

  const float* X      = (const float*)d_in[0];  // (5,64,2,64,64)
  const float* U      = (const float*)d_in[1];  // (4,64,9)
  const float* dt     = (const float*)d_in[3];  // (64,1)
  const float* coords = (const float*)d_in[4];  // (5,64,256,2)

  // -- params (jax tree-leaf order: dict keys sorted alphabetically) --
  int pi = 5;
  auto P = [&]() -> const float* { return (const float*)d_in[pi++]; };
  struct ResP { const float *b1, *b2, *be1, *be2, *g1, *g2, *w1, *w2; };
  // dec
  const float* cf_b = P(); const float* cf_w = P();
  const float *d1b = P(), *d1be = P(), *d1g = P(), *d1w = P();
  const float *d2b = P(), *d2be = P(), *d2g = P(), *d2w = P();
  const float *d3b = P(), *d3be = P(), *d3g = P(), *d3w = P();
  const float *d4b = P(), *d4be = P(), *d4g = P(), *d4w = P();
  const float* dfc_b = P(); const float* dfc_w = P();
  ResP dres[3];
  for (int i = 0; i < 3; ++i) {
    dres[i].b1 = P(); dres[i].b2 = P(); dres[i].be1 = P(); dres[i].be2 = P();
    dres[i].g1 = P(); dres[i].g2 = P(); dres[i].w1 = P(); dres[i].w2 = P();
  }
  // enc
  const float *c1b = P(), *c1be = P(), *c1g = P(), *c1w = P();
  const float *c2b = P(), *c2be = P(), *c2g = P(), *c2w = P();
  const float *c3b = P(), *c3be = P(), *c3g = P(), *c3w = P();
  const float *c4b = P(), *c4be = P(), *c4g = P(), *c4w = P();
  const float* fcd_b = P(); const float* fcd_w = P();
  const float* fcp_b = P(); const float* fcp_w = P();
  ResP eres[3];
  for (int i = 0; i < 3; ++i) {
    eres[i].b1 = P(); eres[i].b2 = P(); eres[i].be1 = P(); eres[i].be2 = P();
    eres[i].g1 = P(); eres[i].g2 = P(); eres[i].w1 = P(); eres[i].w2 = P();
  }
  // pinn
  const float *pb1 = P(), *pb2 = P(), *pb3 = P(), *pb4 = P();
  const float *pw1 = P(), *pw2 = P(), *pw3 = P(), *pw4 = P();
  // tr
  const float *A_b = P(), *A_w = P(), *B_b = P(), *B_w = P();
  const float *C_b = P(), *C_w = P(), *D_b = P(), *D_w = P();
  const float *te_b[3], *te_be[3], *te_g[3], *te_w[3];
  for (int i = 0; i < 3; ++i) { te_b[i] = P(); te_be[i] = P(); te_g[i] = P(); te_w[i] = P(); }

  // -- workspace arena --
  char* ap = (char*)d_ws;
  auto alloc = [&](size_t bytes) -> void* {
    void* r = (void*)ap; ap += (bytes + 255) & ~(size_t)255; return r;
  };
  const long MAXACT = 20971520;                 // 320*16*64*64 floats
  half_t* col  = (half_t*)alloc((size_t)1310720 * 192 * 2);  // largest padded im2col
  float*  bA   = (float*)alloc((size_t)MAXACT * 4);
  float*  bB   = (float*)alloc((size_t)MAXACT * 4);
  float*  bT   = (float*)alloc((size_t)MAXACT * 4);
  half_t* cvt16 = (half_t*)alloc((size_t)320 * 32768 * 2);   // f16 GEMM-A staging
  float*  meanb = (float*)alloc(4096 * 4);
  float*  rstdb = (float*)alloc(4096 * 4);
  float*  tdms  = (float*)alloc(256 * 4);                    // TDM probe sink
  float*  zd    = (float*)alloc((size_t)320 * 64 * 4);
  float*  zp    = (float*)alloc((size_t)320 * 32 * 4);
  float*  zall  = (float*)alloc((size_t)320 * 96 * 4);
  float*  zsmat = (float*)alloc((size_t)320 * 96 * 4);
  float*  Zp    = (float*)alloc((size_t)256 * 96 * 4);
  float*  Yp    = (float*)alloc((size_t)4 * 64 * 14 * 4);
  float*  Atb   = (float*)alloc((size_t)64 * 9216 * 4);
  float*  Btb   = (float*)alloc((size_t)64 * 864 * 4);
  float*  Ctb   = (float*)alloc((size_t)64 * 896 * 4);
  float*  Dtb   = (float*)alloc((size_t)64 * 126 * 4);
  float*  tf0   = (float*)alloc((size_t)64 * 200 * 4);
  float*  tf1   = (float*)alloc((size_t)64 * 200 * 4);
  half_t* hz16  = (half_t*)alloc((size_t)64 * 128 * 2);
  // f16 weights [N][Kp]
  half_t* wc1 = (half_t*)alloc((size_t)16 * 64 * 2);
  half_t* wc2 = (half_t*)alloc((size_t)32 * 192 * 2);
  half_t* wc3 = (half_t*)alloc((size_t)64 * 320 * 2);
  half_t* wc4 = (half_t*)alloc((size_t)128 * 576 * 2);
  half_t *ew1[3], *ew2[3], *dw1[3], *dw2[3];
  for (int i = 0; i < 3; ++i) {
    ew1[i] = (half_t*)alloc((size_t)128 * 1152 * 2);
    ew2[i] = (half_t*)alloc((size_t)128 * 1152 * 2);
    dw1[i] = (half_t*)alloc((size_t)128 * 1152 * 2);
    dw2[i] = (half_t*)alloc((size_t)128 * 1152 * 2);
  }
  half_t* wfcd = (half_t*)alloc((size_t)64 * 32768 * 2);
  half_t* wfcp = (half_t*)alloc((size_t)32 * 32768 * 2);
  half_t* wdfc = (half_t*)alloc((size_t)32768 * 128 * 2);
  half_t* wd1 = (half_t*)alloc((size_t)64 * 1152 * 2);
  half_t* wd2 = (half_t*)alloc((size_t)32 * 256 * 2);
  half_t* wd3 = (half_t*)alloc((size_t)16 * 320 * 2);
  half_t* wd4 = (half_t*)alloc((size_t)16 * 64 * 2);
  half_t* wcf = (half_t*)alloc((size_t)2 * 192 * 2);
  half_t* pw1t = (half_t*)alloc((size_t)128 * 64 * 2);
  half_t* pw2t = (half_t*)alloc((size_t)128 * 128 * 2);
  half_t* pw3t = (half_t*)alloc((size_t)128 * 128 * 2);
  half_t* pw4t = (half_t*)alloc((size_t)2 * 128 * 2);
  half_t* tew[3];
  tew[0] = (half_t*)alloc((size_t)200 * 128 * 2);
  tew[1] = (half_t*)alloc((size_t)200 * 256 * 2);
  tew[2] = (half_t*)alloc((size_t)96 * 256 * 2);
  half_t* Awt = (half_t*)alloc((size_t)9216 * 128 * 2);
  half_t* Bwt = (half_t*)alloc((size_t)864 * 128 * 2);
  half_t* Cwt = (half_t*)alloc((size_t)896 * 128 * 2);
  half_t* Dwt = (half_t*)alloc((size_t)126 * 128 * 2);

  // -- weight prep (all [N][Kp], no transpose) --
  prep(s, c1w, wc1, 16, 18, 64);    prep(s, c2w, wc2, 32, 144, 192);
  prep(s, c3w, wc3, 64, 288, 320);  prep(s, c4w, wc4, 128, 576, 576);
  for (int i = 0; i < 3; ++i) {
    prep(s, eres[i].w1, ew1[i], 128, 1152, 1152);
    prep(s, eres[i].w2, ew2[i], 128, 1152, 1152);
    prep(s, dres[i].w1, dw1[i], 128, 1152, 1152);
    prep(s, dres[i].w2, dw2[i], 128, 1152, 1152);
  }
  prep(s, fcd_w, wfcd, 64, 32768, 32768);
  prep(s, fcp_w, wfcp, 32, 32768, 32768);
  prep(s, dfc_w, wdfc, 32768, 96, 128);
  k_prep_deconv<<<dim3(1024), dim3(256), 0, s>>>(d1w, wd1, 128, 64, 3, 1152);
  k_prep_deconv<<<dim3(1024), dim3(256), 0, s>>>(d2w, wd2, 64, 32, 2, 256);
  k_prep_deconv<<<dim3(1024), dim3(256), 0, s>>>(d3w, wd3, 32, 16, 3, 320);
  k_prep_deconv<<<dim3(1024), dim3(256), 0, s>>>(d4w, wd4, 16, 16, 2, 64);
  prep(s, cf_w, wcf, 2, 144, 192);
  prep(s, pw1, pw1t, 128, 34, 64);   prep(s, pw2, pw2t, 128, 128, 128);
  prep(s, pw3, pw3t, 128, 128, 128); prep(s, pw4, pw4t, 2, 128, 128);
  prep(s, te_w[0], tew[0], 200, 97, 128);
  prep(s, te_w[1], tew[1], 200, 200, 256);
  prep(s, te_w[2], tew[2], 96, 200, 256);
  prep(s, A_w, Awt, 9216, 96, 128); prep(s, B_w, Bwt, 864, 96, 128);
  prep(s, C_w, Cwt, 896, 96, 128);  prep(s, D_w, Dwt, 126, 96, 128);

  // TDM path probe on a real weight tile (self-contained, sink only)
  k_tdm_probe<<<dim3(1), dim3(32), 0, s>>>(wc4, tdms, 576, 128);

  // ================= encoder (320 images) =================
  conv_layer(s, X, col, wc1, c1b, bT, 320, 2, 64, 64, 3, 1, 2, 1, 16);
  bn_relu(s, bT, nullptr, bA, c1g, c1be, meanb, rstdb, 5, 64, 16, 1024, 1);
  conv_layer(s, bA, col, wc2, c2b, bT, 320, 16, 32, 32, 3, 1, 1, 1, 32);
  bn_relu(s, bT, nullptr, bB, c2g, c2be, meanb, rstdb, 5, 64, 32, 1024, 1);
  conv_layer(s, bB, col, wc3, c3b, bT, 320, 32, 32, 32, 3, 1, 2, 1, 64);
  bn_relu(s, bT, nullptr, bA, c3g, c3be, meanb, rstdb, 5, 64, 64, 256, 1);
  conv_layer(s, bA, col, wc4, c4b, bT, 320, 64, 16, 16, 3, 1, 1, 1, 128);
  bn_relu(s, bT, nullptr, bB, c4g, c4be, meanb, rstdb, 5, 64, 128, 256, 1);
  for (int i = 0; i < 3; ++i) {  // residual blocks, x lives in bB
    conv_layer(s, bB, col, ew1[i], eres[i].b1, bT, 320, 128, 16, 16, 3, 1, 1, 1, 128);
    bn_relu(s, bT, nullptr, bA, eres[i].g1, eres[i].be1, meanb, rstdb, 5, 64, 128, 256, 1);
    conv_layer(s, bA, col, ew2[i], eres[i].b2, bT, 320, 128, 16, 16, 3, 1, 1, 1, 128);
    bn_relu(s, bT, bB, bB, eres[i].g2, eres[i].be2, meanb, rstdb, 5, 64, 128, 256, 1);
  }
  k_cvt_pad<<<dim3(4096), dim3(256), 0, s>>>(bB, cvt16, 320, 32768, 32768);
  run_gemm(s, cvt16, wfcd, fcd_b, zd, 320, 64, 32768, 0, 0);
  run_gemm(s, cvt16, wfcp, fcp_b, zp, 320, 32, 32768, 0, 0);
  k_concat_z<<<dim3(128), dim3(256), 0, s>>>(zd, zp, zall, 320);

  // ================= transition =================
  k_build_hz<<<dim3(32), dim3(256), 0, s>>>(zall, dt, hz16, 64);  // rows 0..63 = t0
  run_gemm(s, hz16, tew[0], te_b[0], tf0, 64, 200, 128, 0, 0);
  bn_relu(s, tf0, nullptr, tf1, te_g[0], te_be[0], meanb, rstdb, 1, 64, 200, 1, 1);
  k_cvt_pad<<<dim3(64), dim3(256), 0, s>>>(tf1, cvt16, 64, 200, 256);
  run_gemm(s, cvt16, tew[1], te_b[1], tf0, 64, 200, 256, 0, 0);
  bn_relu(s, tf0, nullptr, tf1, te_g[1], te_be[1], meanb, rstdb, 1, 64, 200, 1, 1);
  k_cvt_pad<<<dim3(64), dim3(256), 0, s>>>(tf1, cvt16, 64, 200, 256);
  run_gemm(s, cvt16, tew[2], te_b[2], tf0, 64, 96, 256, 0, 0);
  bn_relu(s, tf0, nullptr, tf1, te_g[2], te_be[2], meanb, rstdb, 1, 64, 96, 1, 1);
  k_cvt_pad<<<dim3(64), dim3(256), 0, s>>>(tf1, cvt16, 64, 96, 128);
  run_gemm(s, cvt16, Awt, A_b, Atb, 64, 9216, 128, 0, 0);
  run_gemm(s, cvt16, Bwt, B_b, Btb, 64, 864, 128, 0, 0);
  run_gemm(s, cvt16, Cwt, C_b, Ctb, 64, 896, 128, 0, 0);
  run_gemm(s, cvt16, Dwt, D_b, Dtb, 64, 126, 128, 0, 0);
  const float* zprev = zall;  // z0 = rows 0..63
  for (int t = 0; t < 4; ++t) {
    k_step<<<dim3(64), dim3(128), 0, s>>>(Atb, Btb, Ctb, Dtb, U + (size_t)t * 64 * 9,
                                          dt, zprev, Zp + (size_t)t * 64 * 96,
                                          Yp + (size_t)t * 64 * 14);
    zprev = Zp + (size_t)t * 64 * 96;
  }

  // ================= decoder (320 latents) =================
  hipMemcpyAsync(zsmat, zall, (size_t)64 * 96 * 4, hipMemcpyDeviceToDevice, s);
  hipMemcpyAsync(zsmat + 64 * 96, Zp, (size_t)256 * 96 * 4, hipMemcpyDeviceToDevice, s);
  k_cvt_pad<<<dim3(128), dim3(256), 0, s>>>(zsmat, cvt16, 320, 96, 128);
  run_gemm(s, cvt16, wdfc, dfc_b, bA, 320, 32768, 128, 0, 1);  // fc + relu
  for (int i = 0; i < 3; ++i) {  // residual blocks, x lives in bA
    conv_layer(s, bA, col, dw1[i], dres[i].b1, bT, 320, 128, 16, 16, 3, 1, 1, 1, 128);
    bn_relu(s, bT, nullptr, bB, dres[i].g1, dres[i].be1, meanb, rstdb, 5, 64, 128, 256, 1);
    conv_layer(s, bB, col, dw2[i], dres[i].b2, bT, 320, 128, 16, 16, 3, 1, 1, 1, 128);
    bn_relu(s, bT, bA, bA, dres[i].g2, dres[i].be2, meanb, rstdb, 5, 64, 128, 256, 1);
  }
  conv_layer(s, bA, col, wd1, d1b, bT, 320, 128, 16, 16, 3, 1, 1, 1, 64);
  bn_relu(s, bT, nullptr, bB, d1g, d1be, meanb, rstdb, 5, 64, 64, 256, 1);
  conv_layer(s, bB, col, wd2, d2b, bT, 320, 64, 16, 16, 2, 1, 1, 2, 32);
  bn_relu(s, bT, nullptr, bA, d2g, d2be, meanb, rstdb, 5, 64, 32, 1024, 1);
  conv_layer(s, bA, col, wd3, d3b, bT, 320, 32, 32, 32, 3, 1, 1, 1, 16);
  bn_relu(s, bT, nullptr, bB, d3g, d3be, meanb, rstdb, 5, 64, 16, 1024, 1);
  conv_layer(s, bB, col, wd4, d4b, bT, 320, 16, 32, 32, 2, 1, 1, 2, 16);
  bn_relu(s, bT, nullptr, bA, d4g, d4be, meanb, rstdb, 5, 64, 16, 4096, 1);
  // final conv writes straight into d_out[0 .. 2621440): x0_rec ++ X_next_pred
  conv_layer(s, bA, col, wcf, cf_b, (float*)d_out, 320, 16, 64, 64, 3, 1, 1, 1, 2);

  // ================= PINN =================
  const long OFF_ZP = 2621440, OFF_ZN = 2646016, OFF_YP = 2670592;
  const long OFF_PN = 2674176, OFF_T0 = 2805248;
  float* ph0 = bT;  // act buffers free now
  float* ph1 = bB;
  k_pinn_in<<<dim3(2048), dim3(256), 0, s>>>(Zp, 96, 64, coords, cvt16, (long)4 * 64 * 256);
  run_gemm(s, cvt16, pw1t, pb1, ph0, 65536, 128, 64, 0, 2);
  k_cvt_pad<<<dim3(4096), dim3(256), 0, s>>>(ph0, cvt16, 65536, 128, 128);
  run_gemm(s, cvt16, pw2t, pb2, ph1, 65536, 128, 128, 0, 2);
  k_cvt_pad<<<dim3(4096), dim3(256), 0, s>>>(ph1, cvt16, 65536, 128, 128);
  run_gemm(s, cvt16, pw3t, pb3, ph0, 65536, 128, 128, 0, 2);
  k_cvt_pad<<<dim3(4096), dim3(256), 0, s>>>(ph0, cvt16, 65536, 128, 128);
  run_gemm(s, cvt16, pw4t, pb4, (float*)d_out + OFF_PN, 65536, 2, 128, 0, 0);
  // pinn_t0: z0p = zp rows 0..63, coords[T]
  k_pinn_in<<<dim3(1024), dim3(256), 0, s>>>(zp, 32, 0, coords + (size_t)4 * 64 * 256 * 2,
                                             cvt16, (long)64 * 256);
  run_gemm(s, cvt16, pw1t, pb1, ph0, 16384, 128, 64, 0, 2);
  k_cvt_pad<<<dim3(2048), dim3(256), 0, s>>>(ph0, cvt16, 16384, 128, 128);
  run_gemm(s, cvt16, pw2t, pb2, ph1, 16384, 128, 128, 0, 2);
  k_cvt_pad<<<dim3(2048), dim3(256), 0, s>>>(ph1, cvt16, 16384, 128, 128);
  run_gemm(s, cvt16, pw3t, pb3, ph0, 16384, 128, 128, 0, 2);
  k_cvt_pad<<<dim3(2048), dim3(256), 0, s>>>(ph0, cvt16, 16384, 128, 128);
  run_gemm(s, cvt16, pw4t, pb4, (float*)d_out + OFF_T0, 16384, 2, 128, 0, 0);

  // ================= remaining outputs =================
  hipMemcpyAsync((float*)d_out + OFF_ZP, Zp, (size_t)256 * 96 * 4,
                 hipMemcpyDeviceToDevice, s);
  hipMemcpyAsync((float*)d_out + OFF_ZN, zall + 64 * 96, (size_t)256 * 96 * 4,
                 hipMemcpyDeviceToDevice, s);
  hipMemcpyAsync((float*)d_out + OFF_YP, Yp, (size_t)4 * 64 * 14 * 4,
                 hipMemcpyDeviceToDevice, s);
}
